// _GptOssDecoderLayer_85787676770794
// MI455X (gfx1250) — matrix operand order repaired and checked by reading, not compiled, not run on hardware
//
#include <hip/hip_runtime.h>
#include <hip/hip_bf16.h>

// ---------------------------------------------------------------------------
// GPT-OSS decoder layer for MI455X (gfx1250), wave32 + WMMA bf16.
//   B=1, S=1024, H=2048, NH=32, NKV=8, DH=64 (GQA G=4), E=8, K=2, I=2048
// All big GEMMs run on v_wmma_f32_16x16x32_bf16 (bf16 in, f32 accumulate).
// fp32 weights are converted to bf16 in the B-tile loader (packed cvt when
// available); activations are produced as bf16 by each producer's epilogue.
// ---------------------------------------------------------------------------

static constexpr int S_   = 1024;
static constexpr int H_   = 2048;
static constexpr int NH_  = 32;
static constexpr int NKV_ = 8;
static constexpr int DH_  = 64;
static constexpr int GQ_  = 4;      // NH / NKV
static constexpr int E_   = 8;
static constexpr int I_   = 2048;
static constexpr int NHD_ = NH_ * DH_;   // 2048

typedef __attribute__((ext_vector_type(16))) __bf16  v16bf;
typedef __attribute__((ext_vector_type(2)))  __bf16  v2bf;
typedef __attribute__((ext_vector_type(8)))  float   v8f;
typedef __attribute__((ext_vector_type(4)))  unsigned int u32x4;
typedef __attribute__((ext_vector_type(4)))  float   f32x4;

#define DEV static __device__ __forceinline__

DEV __bf16 f2bf(float f) {
    unsigned int u = __builtin_bit_cast(unsigned int, f);
    unsigned int r = u + 0x7FFFu + ((u >> 16) & 1u);   // round-to-nearest-even
    unsigned short s = (unsigned short)(r >> 16);
    return __builtin_bit_cast(__bf16, s);
}
DEV float bf2f(__bf16 b) {
    unsigned short s = __builtin_bit_cast(unsigned short, b);
    unsigned int u = ((unsigned int)s) << 16;
    return __builtin_bit_cast(float, u);
}
DEV v2bf pack2bf(float lo, float hi) {
#if __has_builtin(__builtin_amdgcn_cvt_pk_bf16_f32)
    return __builtin_amdgcn_cvt_pk_bf16_f32(lo, hi);
#else
    v2bf r; r[0] = f2bf(lo); r[1] = f2bf(hi); return r;
#endif
}

DEV v8f wmma_bf16(v16bf a, v16bf b, v8f c) {
    // D = A(16x32) * B(32x16) + C, f32 accumulate
    return __builtin_amdgcn_wmma_f32_16x16x32_bf16(
        /*neg_a=*/false, a, /*neg_b=*/false, b,
        /*c_mod=*/(short)0, c, /*reuse_a=*/false, /*reuse_b=*/false);
}

union V16U  { u32x4 u[2]; v16bf v; };
union V16P  { v2bf  h[8]; v16bf v; };

// A operand: 16 rows (M = lane&15), K contiguous:
// lanes<16 hold K k0..k0+15; lanes>=16 hold K k0+16..k0+31
DEV v16bf load_tileA_bf16(const __bf16* base, int ld, int k0, int lane) {
    int m = lane & 15, koff = (lane >> 4) << 4;
    const __bf16* p = base + (size_t)m * ld + k0 + koff;
    V16U t;
    t.u[0] = *(const u32x4*)p;          // K koff..koff+7
    t.u[1] = *(const u32x4*)(p + 8);    // K koff+8..koff+15
    return t.v;
}

// B operand: 16 cols (N = lane&15); lanes<16 hold K k0..k0+15 sequential,
// lanes>=16 hold K k0+16..k0+31 (same contiguous k order as load_tileA_bf16).
DEV v16bf load_tileB_bf16(const __bf16* base, int ld, int k0, int lane) {
    int n = lane & 15, koff = (lane >> 4) << 4;
    const __bf16* p = base + (size_t)n * ld + k0 + koff;
    V16U t;
    t.u[0] = *(const u32x4*)p;
    t.u[1] = *(const u32x4*)(p + 8);
    return t.v;
}

// B operand sourced from fp32 weights ([N,K] row major), converted inline.
DEV v16bf load_tileB_f32(const float* base, int ld, int k0, int lane) {
    int n = lane & 15, koff = (lane >> 4) << 4;
    const float* p = base + (size_t)n * ld + k0 + koff;
    f32x4 f0 = *(const f32x4*)(p);
    f32x4 f1 = *(const f32x4*)(p + 4);
    f32x4 f2 = *(const f32x4*)(p + 8);
    f32x4 f3 = *(const f32x4*)(p + 12);
    V16P t;
    t.h[0] = pack2bf(f0[0], f0[1]);
    t.h[1] = pack2bf(f0[2], f0[3]);
    t.h[2] = pack2bf(f1[0], f1[1]);
    t.h[3] = pack2bf(f1[2], f1[3]);
    t.h[4] = pack2bf(f2[0], f2[1]);
    t.h[5] = pack2bf(f2[2], f2[3]);
    t.h[6] = pack2bf(f3[0], f3[1]);
    t.h[7] = pack2bf(f3[2], f3[3]);
    return t.v;
}

// ---------------------------------------------------------------------------
// RMSNorm (fp32 in) + cast to bf16. grid = S rows, 256 threads.
// ---------------------------------------------------------------------------
__global__ __launch_bounds__(256)
void rmsnorm_cast_kernel(const float* __restrict__ x, const float* __restrict__ g,
                         __bf16* __restrict__ out, int H) {
    int row = blockIdx.x, tid = threadIdx.x;
    const float* xr = x + (size_t)row * H;
    float ss = 0.f;
    for (int i = tid; i < H; i += 256) { float v = xr[i]; ss += v * v; }
    __shared__ float red[256];
    red[tid] = ss; __syncthreads();
    for (int s = 128; s > 0; s >>= 1) {
        if (tid < s) red[tid] += red[tid + s];
        __syncthreads();
    }
    float inv = rsqrtf(red[0] / (float)H + 1e-5f);
    for (int i = tid; i < H; i += 256)
        out[(size_t)row * H + i] = f2bf(xr[i] * inv * g[i]);
}

// ---------------------------------------------------------------------------
// Router: logits = hn @ router_w^T + b, top-2 softmax -> dense wts[S,E]
// grid = S, block = 256 (8 waves; wave e computes logit e).
// ---------------------------------------------------------------------------
__global__ __launch_bounds__(256)
void router_kernel(const __bf16* __restrict__ hn, const float* __restrict__ rw,
                   const float* __restrict__ rb, float* __restrict__ wts) {
    int s = blockIdx.x, tid = threadIdx.x;
    int e = tid >> 5, l = tid & 31;
    const __bf16* hr = hn + (size_t)s * H_;
    const float*  wr = rw + (size_t)e * H_;
    float acc = 0.f;
    for (int i = l; i < H_; i += 32) acc += bf2f(hr[i]) * wr[i];
#pragma unroll
    for (int off = 16; off > 0; off >>= 1) acc += __shfl_xor(acc, off, 32);
    __shared__ float lg[E_];
    if (l == 0) lg[e] = acc + rb[e];
    __syncthreads();
    if (tid == 0) {
        int i0 = 0;
        for (int i = 1; i < E_; ++i) if (lg[i] > lg[i0]) i0 = i;
        int i1 = -1;
        for (int i = 0; i < E_; ++i) {
            if (i == i0) continue;
            if (i1 < 0 || lg[i] > lg[i1]) i1 = i;
        }
        float w0 = 1.f / (1.f + __expf(lg[i1] - lg[i0]));
        for (int i = 0; i < E_; ++i) wts[(size_t)s * E_ + i] = 0.f;
        wts[(size_t)s * E_ + i0] = w0;
        wts[(size_t)s * E_ + i1] = 1.f - w0;
    }
}

// ---------------------------------------------------------------------------
// Generic WMMA GEMM: C[M,N] = A[M,K](bf16) @ Bw[N,K](f32)^T, epilogue by EPI.
// Block = 256 (8 waves as 2x4), block tile 64x256, wave tile 32x64:
// 8 WMMAs per K-step on 6 tile loads. fp32 weight stream is prefetched
// (global_prefetch_b8) two K-steps ahead.
//  EPI 0: q/k proj: bf16 store head-major [Nh, S, 64]       (+bias)
//  EPI 1: v proj:   bf16 store transposed [Nkv, 64, S]      (+bias)
//  EPI 2: Wo:       f32 out = acc + bias + resid            (residual add)
//  EPI 3: gate:     f32 out = acc + bias          (expert skip via wts)
//  EPI 4: up+glu:   reads gate f32, writes gated bf16       (expert skip)
//  EPI 5: down:     f32 out += wts[m,e] * (acc + bias)      (expert skip)
// ---------------------------------------------------------------------------
template <int EPI>
__global__ __launch_bounds__(256)
void gemm_kernel(const __bf16* __restrict__ A, const float* __restrict__ Bw,
                 const float* __restrict__ bias, int M, int N, int Kd,
                 float* __restrict__ outF, __bf16* __restrict__ outB,
                 const float* __restrict__ resid, const float* __restrict__ aux,
                 const float* __restrict__ wts, int expert, int Sdim) {
    int tid = threadIdx.x, lane = tid & 31, w = tid >> 5;
    int m0 = blockIdx.y * 64 + (w >> 2) * 32;
    int n0 = blockIdx.x * 256 + (w & 3) * 64;

    if constexpr (EPI >= 3) {
        __shared__ int sflag;
        if (tid == 0) sflag = 0;
        __syncthreads();
        int mb = blockIdx.y * 64;
        if (tid < 64 && wts[(size_t)(mb + tid) * E_ + expert] != 0.0f) sflag = 1;
        __syncthreads();
        if (!sflag) return;   // no token in this 64-row group routed to expert
    }

    v8f acc[2][4] = {};
    for (int k0 = 0; k0 < Kd; k0 += 32) {
        // prefetch fp32 weight stream 2 K-steps ahead (64 rows x 1 line)
        if (k0 + 64 < Kd) {
            __builtin_prefetch(Bw + (size_t)(n0 + lane) * Kd + k0 + 64, 0, 1);
            __builtin_prefetch(Bw + (size_t)(n0 + 32 + lane) * Kd + k0 + 64, 0, 1);
        }
        v16bf a0 = load_tileA_bf16(A + (size_t)m0 * Kd, Kd, k0, lane);
        v16bf a1 = load_tileA_bf16(A + (size_t)(m0 + 16) * Kd, Kd, k0, lane);
        v16bf b[4];
#pragma unroll
        for (int j = 0; j < 4; ++j)
            b[j] = load_tileB_f32(Bw + (size_t)(n0 + 16 * j) * Kd, Kd, k0, lane);
#pragma unroll
        for (int j = 0; j < 4; ++j) {
            acc[0][j] = wmma_bf16(a0, b[j], acc[0][j]);
            acc[1][j] = wmma_bf16(a1, b[j], acc[1][j]);
        }
    }

    int nl = lane & 15, mh = (lane >> 4) << 3;
#pragma unroll
    for (int i = 0; i < 2; ++i)
#pragma unroll
        for (int j = 0; j < 4; ++j)
#pragma unroll
            for (int r = 0; r < 8; ++r) {
                int m = m0 + i * 16 + mh + r;
                int n = n0 + j * 16 + nl;
                float v = acc[i][j][r] + bias[n];
                if constexpr (EPI == 0) {        // [n/64, S, 64] bf16
                    outB[((size_t)(n >> 6) * Sdim + m) * 64 + (n & 63)] = f2bf(v);
                } else if constexpr (EPI == 1) { // [n/64, 64, S] bf16 (v^T)
                    outB[(size_t)(n >> 6) * 64 * Sdim + (size_t)(n & 63) * Sdim + m] = f2bf(v);
                } else if constexpr (EPI == 2) {
                    outF[(size_t)m * N + n] = v + resid[(size_t)m * N + n];
                } else if constexpr (EPI == 3) {
                    outF[(size_t)m * N + n] = v;
                } else if constexpr (EPI == 4) {
                    float g  = aux[(size_t)m * N + n];
                    float gc = fminf(g, 7.0f);
                    float uc = fminf(fmaxf(v, -7.0f), 7.0f);
                    float glu = gc / (1.0f + __expf(-1.702f * gc));
                    outB[(size_t)m * N + n] = f2bf((uc + 1.0f) * glu);
                } else if constexpr (EPI == 5) {
                    outF[(size_t)m * N + n] += wts[(size_t)m * E_ + expert] * v;
                }
            }
}

// ---------------------------------------------------------------------------
// Flash attention with sink, transposed (scores^T = K * Q^T), 1 wave per
// (16-query block, head). Sink enters online-softmax initial state only.
// All K- and V-tiles of a 32-key block are issued before any WMMA so the
// b128 loads pipeline and V loads overlap the softmax VALU work.
// ---------------------------------------------------------------------------
__global__ __launch_bounds__(32)
void attn_kernel(const __bf16* __restrict__ qh,  // [NH, S, DH] bf16
                 const __bf16* __restrict__ kh,  // [NKV, S, DH] bf16
                 const __bf16* __restrict__ vT,  // [NKV, DH, S] bf16
                 const float* __restrict__ sinks,
                 __bf16* __restrict__ attn_out)  // [S, NH*DH] bf16
{
    const int lane = threadIdx.x;
    const int q0 = blockIdx.x * 16;
    const int h = blockIdx.y;
    const int kvh = h / GQ_;
    const float scale = 0.125f;                 // DH^-0.5

    v16bf bq[2];                                // q^T as B operand, K = d
    bq[0] = load_tileB_bf16(qh + (size_t)(h * S_ + q0) * DH_, DH_, 0, lane);
    bq[1] = load_tileB_bf16(qh + (size_t)(h * S_ + q0) * DH_, DH_, 32, lane);

    float m_run = sinks[h];                     // sink seeds max
    float l_run = 1.0f;                         // exp(sink - m_run)
    v8f acc[4] = {};                            // attn^T, 4 d-chunks x 16 q

    const int qg = q0 + (lane & 15);
    const int kend = q0 + 16;
    for (int k0 = 0; k0 < kend; k0 += 32) {
        // ---- issue ALL loads for this key block up front ----
        v16bf ak[2][2];                         // [key subtile][dstep]
#pragma unroll
        for (int ds = 0; ds < 2; ++ds) {
            ak[0][ds] = load_tileA_bf16(kh + (size_t)(kvh * S_ + k0) * DH_, DH_, ds * 32, lane);
            ak[1][ds] = load_tileA_bf16(kh + (size_t)(kvh * S_ + k0 + 16) * DH_, DH_, ds * 32, lane);
        }
        v16bf av[4];                            // v^T d-chunks, K = 32 keys
#pragma unroll
        for (int c = 0; c < 4; ++c)
            av[c] = load_tileA_bf16(vT + (size_t)(kvh * DH_ + c * 16) * S_, S_, k0, lane);

        // ---- scores^T = K * Q^T ----
        v8f s0 = {}, s1 = {};
#pragma unroll
        for (int ds = 0; ds < 2; ++ds) {
            s0 = wmma_bf16(ak[0][ds], bq[ds], s0);
            s1 = wmma_bf16(ak[1][ds], bq[ds], s1);
        }

        // ---- scale + causal mask; per-query online softmax ----
        float p0[8], p1[8];
        int krb = k0 + ((lane >> 4) << 3);
        float mx = -3.0e38f;
#pragma unroll
        for (int r = 0; r < 8; ++r) {
            float x0 = (krb + r      <= qg) ? s0[r] * scale : -1.0e30f;
            float x1 = (krb + 16 + r <= qg) ? s1[r] * scale : -1.0e30f;
            p0[r] = x0; p1[r] = x1;
            mx = fmaxf(mx, fmaxf(x0, x1));
        }
        mx = fmaxf(mx, __shfl_xor(mx, 16, 32));     // combine lane pair (same q)
        float m_new = fmaxf(m_run, mx);
        float sum = 0.f;
#pragma unroll
        for (int r = 0; r < 8; ++r) {
            p0[r] = __expf(p0[r] - m_new);
            p1[r] = __expf(p1[r] - m_new);
            sum += p0[r] + p1[r];
        }
        sum += __shfl_xor(sum, 16, 32);
        float corr = __expf(m_run - m_new);
        l_run = l_run * corr + sum;
        m_run = m_new;
#pragma unroll
        for (int c = 0; c < 4; ++c)
#pragma unroll
            for (int r = 0; r < 8; ++r) acc[c][r] *= corr;

        // ---- repack P^T (C layout) into B layout: one lane-pair exchange ----
        float lo[8], hi[8];
#pragma unroll
        for (int r = 0; r < 8; ++r) {
            float p0x = __shfl_xor(p0[r], 16, 32);
            float p1x = __shfl_xor(p1[r], 16, 32);
            bool up = lane >= 16;
            lo[r] = up ? p1x   : p0[r];   // keys 0..7   | 16..23
            hi[r] = up ? p1[r] : p0x;     // keys 8..15  | 24..31
        }
        V16P bp;
#pragma unroll
        for (int r = 0; r < 4; ++r) {
            bp.h[r]     = pack2bf(lo[2 * r], lo[2 * r + 1]);
            bp.h[4 + r] = pack2bf(hi[2 * r], hi[2 * r + 1]);
        }

        // ---- acc^T += v^T * P^T ----
#pragma unroll
        for (int c = 0; c < 4; ++c)
            acc[c] = wmma_bf16(av[c], bp.v, acc[c]);
    }

    float inv = 1.0f / l_run;
    int dbase = ((lane >> 4) << 3);
#pragma unroll
    for (int c = 0; c < 4; ++c) {
        union { v2bf h[4]; u32x4 u; } pk;
#pragma unroll
        for (int r = 0; r < 4; ++r)
            pk.h[r] = pack2bf(acc[c][2 * r] * inv, acc[c][2 * r + 1] * inv);
        __bf16* dst = attn_out + (size_t)(q0 + (lane & 15)) * NHD_
                    + h * DH_ + c * 16 + dbase;
        *(u32x4*)dst = pk.u;
    }
}

// ---------------------------------------------------------------------------
extern "C" void kernel_launch(void* const* d_in, const int* in_sizes, int n_in,
                              void* d_out, int out_size, void* d_ws, size_t ws_size,
                              hipStream_t stream) {
    (void)in_sizes; (void)n_in; (void)out_size; (void)ws_size;
    const float* x        = (const float*)d_in[0];
    // d_in[1] = attention_bias (causal mask) -- recomputed analytically
    const float* ln1_g    = (const float*)d_in[2];
    const float* ln2_g    = (const float*)d_in[3];
    const float* Wq       = (const float*)d_in[4];
    const float* bq       = (const float*)d_in[5];
    const float* Wk       = (const float*)d_in[6];
    const float* bk       = (const float*)d_in[7];
    const float* Wv       = (const float*)d_in[8];
    const float* bv       = (const float*)d_in[9];
    const float* Wo       = (const float*)d_in[10];
    const float* bo       = (const float*)d_in[11];
    const float* sinks    = (const float*)d_in[12];
    const float* router_w = (const float*)d_in[13];
    const float* router_b = (const float*)d_in[14];
    const float* We_gate  = (const float*)d_in[15];
    const float* be_gate  = (const float*)d_in[16];
    const float* We_up    = (const float*)d_in[17];
    const float* be_up    = (const float*)d_in[18];
    const float* We_down  = (const float*)d_in[19];
    const float* be_down  = (const float*)d_in[20];
    float* out = (float*)d_out;

    char* ws = (char*)d_ws;
    size_t off = 0;
    auto alloc = [&](size_t bytes) {
        size_t o = off;
        off += (bytes + 255) & ~(size_t)255;
        return o;
    };
    __bf16* xn_bf    = (__bf16*)(ws + alloc((size_t)S_ * H_ * 2));
    __bf16* q_bf     = (__bf16*)(ws + alloc((size_t)NH_ * S_ * DH_ * 2));
    __bf16* k_bf     = (__bf16*)(ws + alloc((size_t)NKV_ * S_ * DH_ * 2));
    __bf16* vT_bf    = (__bf16*)(ws + alloc((size_t)NKV_ * DH_ * S_ * 2));
    __bf16* attn_bf  = (__bf16*)(ws + alloc((size_t)S_ * NHD_ * 2));
    __bf16* hn_bf    = (__bf16*)(ws + alloc((size_t)S_ * H_ * 2));
    float*  wts      = (float*)(ws + alloc((size_t)S_ * E_ * 4));
    float*  gate_f   = (float*)(ws + alloc((size_t)S_ * I_ * 4));
    __bf16* gated_bf = (__bf16*)(ws + alloc((size_t)S_ * I_ * 2));

    // 1) xn = rmsnorm(x) -> bf16
    rmsnorm_cast_kernel<<<S_, 256, 0, stream>>>(x, ln1_g, xn_bf, H_);

    // 2) projections (WMMA): q [NH,S,DH], k [NKV,S,DH], v^T [NKV,DH,S]
    gemm_kernel<0><<<dim3(NHD_ / 256, S_ / 64), 256, 0, stream>>>(
        xn_bf, Wq, bq, S_, NHD_, H_, nullptr, q_bf, nullptr, nullptr, nullptr, 0, S_);
    gemm_kernel<0><<<dim3((NKV_ * DH_) / 256, S_ / 64), 256, 0, stream>>>(
        xn_bf, Wk, bk, S_, NKV_ * DH_, H_, nullptr, k_bf, nullptr, nullptr, nullptr, 0, S_);
    gemm_kernel<1><<<dim3((NKV_ * DH_) / 256, S_ / 64), 256, 0, stream>>>(
        xn_bf, Wv, bv, S_, NKV_ * DH_, H_, nullptr, vT_bf, nullptr, nullptr, nullptr, 0, S_);

    // 3) flash attention with sink (WMMA) -> attn [S, NH*DH] bf16
    attn_kernel<<<dim3(S_ / 16, NH_), 32, 0, stream>>>(q_bf, k_bf, vT_bf, sinks, attn_bf);

    // 4) h = x + attn @ Wo^T + bo  -> d_out (f32)
    gemm_kernel<2><<<dim3(H_ / 256, S_ / 64), 256, 0, stream>>>(
        attn_bf, Wo, bo, S_, H_, NHD_, out, nullptr, x, nullptr, nullptr, 0, S_);

    // 5) hn = rmsnorm(h) -> bf16 ; router top-2 -> dense wts[S,E]
    rmsnorm_cast_kernel<<<S_, 256, 0, stream>>>(out, ln2_g, hn_bf, H_);
    router_kernel<<<S_, 256, 0, stream>>>(hn_bf, router_w, router_b, wts);

    // 6) MoE: per expert gate -> up(+GLU) -> down(accumulate into d_out),
    //    64-row blocks skipped when no token routes to the expert.
    for (int e = 0; e < E_; ++e) {
        gemm_kernel<3><<<dim3(I_ / 256, S_ / 64), 256, 0, stream>>>(
            hn_bf, We_gate + (size_t)e * I_ * H_, be_gate + (size_t)e * I_,
            S_, I_, H_, gate_f, nullptr, nullptr, nullptr, wts, e, S_);
        gemm_kernel<4><<<dim3(I_ / 256, S_ / 64), 256, 0, stream>>>(
            hn_bf, We_up + (size_t)e * I_ * H_, be_up + (size_t)e * I_,
            S_, I_, H_, nullptr, gated_bf, nullptr, gate_f, wts, e, S_);
        gemm_kernel<5><<<dim3(H_ / 256, S_ / 64), 256, 0, stream>>>(
            gated_bf, We_down + (size_t)e * H_ * I_, be_down + (size_t)e * H_,
            S_, H_, I_, out, nullptr, nullptr, nullptr, wts, e, S_);
    }
}
